// GATRegressor_80436147520135
// MI455X (gfx1250) — matrix operand order repaired and checked by reading, not compiled, not run on hardware
//
#include <hip/hip_runtime.h>

#define F_IN 128
#define H1 4
#define C 32
#define HC 128  // H1*C
#define NEG_SLOPE 0.2f
#define CDIV(a, b) (((a) + (b) - 1) / (b))

typedef __attribute__((ext_vector_type(16))) __bf16 v16bf;
typedef __attribute__((ext_vector_type(8))) float v8f;

__device__ __forceinline__ float leaky(float x) { return x >= 0.f ? x : NEG_SLOPE * x; }

// order-preserving float<->uint mapping for atomicMax-based segment max
__device__ __forceinline__ unsigned f2ord(float f) {
  unsigned u = __float_as_uint(f);
  return (u & 0x80000000u) ? ~u : (u | 0x80000000u);
}
__device__ __forceinline__ float ord2f(unsigned u) {
  return __uint_as_float((u & 0x80000000u) ? (u & 0x7fffffffu) : ~u);
}

__global__ void fill_u32_kernel(unsigned* p, unsigned v, int n) {
  int i = blockIdx.x * blockDim.x + threadIdx.x;
  if (i < n) p[i] = v;
}
__global__ void fill_f32_kernel(float* p, float v, int n) {
  int i = blockIdx.x * blockDim.x + threadIdx.x;
  if (i < n) p[i] = v;
}

// Repack W[K=128, NC] f32 (row-major) into bf16 WMMA B-fragments.
// Layout: wf[((kt*CT + ct)*32 + lane)*16 + e] = bf16(W[(kt*32 + (lane>>4)*16 + e)*NC + ct*16 + (lane&15)])
template <int NC>
__global__ void repack_w_kernel(const float* __restrict__ W, __bf16* __restrict__ wf) {
  const int CT = NC / 16;
  int idx = blockIdx.x * blockDim.x + threadIdx.x;
  int e = idx & 15;
  int lane = (idx >> 4) & 31;
  int rem = idx >> 9;
  int ct = rem % CT;
  int kt = rem / CT;
  if (kt >= 4) return;
  int k = kt * 32 + (lane >> 4) * 16 + e;
  int col = ct * 16 + (lane & 15);
  wf[idx] = (__bf16)W[(size_t)k * NC + col];
}

// out[M, NC] = A[M, 128] @ W[128, NC]  (W pre-repacked as bf16 fragments)
// One wave -> one 16x16 output tile; 8 waves / block.
template <int NC>
__global__ void wmma_gemm_kernel(const float* __restrict__ A, const __bf16* __restrict__ WF,
                                 float* __restrict__ out, int M) {
  const int CT = NC / 16;  // col tiles
  const int RT = 8 / CT;   // row tiles per block
  int lane = threadIdx.x & 31;
  int wave = threadIdx.x >> 5;
  int colTile = wave % CT;
  int rowTile = blockIdx.x * RT + wave / CT;
  int row0 = rowTile * 16;
  if (row0 >= M) return;
  int l16 = lane & 15;
  int half = lane >> 4;
  int aRow = row0 + l16;
  if (aRow >= M) aRow = M - 1;
  const float* arow = A + (size_t)aRow * F_IN;

  v8f acc = {0.f, 0.f, 0.f, 0.f, 0.f, 0.f, 0.f, 0.f};
#pragma unroll
  for (int kt = 0; kt < 4; ++kt) {
    int k0 = kt * 32;
    // A fragment (16x32 bf16): elems 0..15 -> K=k0+half*16+e
    float4 f0 = *(const float4*)(arow + k0 + half * 16);
    float4 f1 = *(const float4*)(arow + k0 + half * 16 + 4);
    float4 f2 = *(const float4*)(arow + k0 + half * 16 + 8);
    float4 f3 = *(const float4*)(arow + k0 + half * 16 + 12);
    v16bf a;
    a[0] = (__bf16)f0.x;  a[1] = (__bf16)f0.y;  a[2] = (__bf16)f0.z;  a[3] = (__bf16)f0.w;
    a[4] = (__bf16)f1.x;  a[5] = (__bf16)f1.y;  a[6] = (__bf16)f1.z;  a[7] = (__bf16)f1.w;
    a[8] = (__bf16)f2.x;  a[9] = (__bf16)f2.y;  a[10] = (__bf16)f2.z; a[11] = (__bf16)f2.w;
    a[12] = (__bf16)f3.x; a[13] = (__bf16)f3.y; a[14] = (__bf16)f3.z; a[15] = (__bf16)f3.w;
    // B fragment (32x16 bf16), 32B contiguous per lane
    v16bf b = *(const v16bf*)(WF + ((size_t)((kt * CT + colTile) * 32 + lane) << 4));
    acc = __builtin_amdgcn_wmma_f32_16x16x32_bf16(false, a, false, b, (short)0, acc, false, false);
  }
  int col = colTile * 16 + l16;
#pragma unroll
  for (int r = 0; r < 8; ++r) {
    int rr = row0 + half * 8 + r;
    if (rr < M) out[(size_t)rr * NC + col] = acc[r];
  }
}

// a_s[n,h] = dot(h[n,h,:], att_s[h,:]) ; a_d likewise
__global__ void att_coef_kernel(const float* __restrict__ h, const float* __restrict__ att_s,
                                const float* __restrict__ att_d, float* __restrict__ as_,
                                float* __restrict__ ad_, int Nn, int Hh) {
  int idx = blockIdx.x * blockDim.x + threadIdx.x;
  if (idx >= Nn * Hh) return;
  int n = idx / Hh, hh = idx - (idx / Hh) * Hh;
  const float* base = h + (size_t)n * (Hh * C) + hh * C;
  const float* sp = att_s + hh * C;
  const float* dp = att_d + hh * C;
  float ss = 0.f, dd = 0.f;
#pragma unroll
  for (int c = 0; c < C; c += 4) {
    float4 v = *(const float4*)(base + c);
    float4 a = *(const float4*)(sp + c);
    float4 b = *(const float4*)(dp + c);
    ss += v.x * a.x + v.y * a.y + v.z * a.z + v.w * a.w;
    dd += v.x * b.x + v.y * b.y + v.z * b.z + v.w * b.w;
  }
  as_[idx] = ss;
  ad_[idx] = dd;
}

__device__ __forceinline__ void edge_sd(const long long* esrc, const long long* edst, int e,
                                        int E0, int& s, int& d) {
  if (e < E0) { s = (int)esrc[e]; d = (int)edst[e]; }
  else { s = e - E0; d = s; }
}

__global__ void edge_max_kernel(const long long* __restrict__ esrc, const long long* __restrict__ edst,
                                int E0, int Nn, int Hh, const float* __restrict__ as_,
                                const float* __restrict__ ad_, unsigned* __restrict__ m) {
  int idx = blockIdx.x * blockDim.x + threadIdx.x;
  if (idx >= (E0 + Nn) * Hh) return;
  int e = idx / Hh, hh = idx - (idx / Hh) * Hh;
  int s, d;
  edge_sd(esrc, edst, e, E0, s, d);
  float val = leaky(as_[s * Hh + hh] + ad_[d * Hh + hh]);
  atomicMax(&m[d * Hh + hh], f2ord(val));
}

__global__ void edge_exp_kernel(const long long* __restrict__ esrc, const long long* __restrict__ edst,
                                int E0, int Nn, int Hh, const float* __restrict__ as_,
                                const float* __restrict__ ad_, const unsigned* __restrict__ m,
                                float* __restrict__ ex, float* __restrict__ denom) {
  int idx = blockIdx.x * blockDim.x + threadIdx.x;
  if (idx >= (E0 + Nn) * Hh) return;
  int e = idx / Hh, hh = idx - (idx / Hh) * Hh;
  int s, d;
  edge_sd(esrc, edst, e, E0, s, d);
  float val = leaky(as_[s * Hh + hh] + ad_[d * Hh + hh]);
  float ee = __expf(val - ord2f(m[d * Hh + hh]));
  ex[idx] = ee;
  atomicAdd(&denom[d * Hh + hh], ee);
}

__global__ void edge_agg_kernel(const long long* __restrict__ esrc, const long long* __restrict__ edst,
                                int E0, int Nn, int Hh, const float* __restrict__ hsrc,
                                const float* __restrict__ ex, const float* __restrict__ denom,
                                float* __restrict__ agg) {
  int idx = blockIdx.x * blockDim.x + threadIdx.x;
  if (idx >= (E0 + Nn) * Hh) return;
  int e = idx / Hh, hh = idx - (idx / Hh) * Hh;
  int s, d;
  edge_sd(esrc, edst, e, E0, s, d);
  float alpha = ex[idx] / denom[d * Hh + hh];
  const float* hp = hsrc + (size_t)s * (Hh * C) + hh * C;
  float* op = agg + (size_t)d * (Hh * C) + hh * C;
#pragma unroll
  for (int c = 0; c < C; c += 4) {
    float4 v = *(const float4*)(hp + c);
    atomicAdd(op + c + 0, v.x * alpha);
    atomicAdd(op + c + 1, v.y * alpha);
    atomicAdd(op + c + 2, v.z * alpha);
    atomicAdd(op + c + 3, v.w * alpha);
  }
}

__global__ void bias_relu_kernel(float* __restrict__ a, const float* __restrict__ b, int Nn, int F) {
  int idx = blockIdx.x * blockDim.x + threadIdx.x;
  if (idx >= Nn * F) return;
  int f = idx % F;
  float v = a[idx] + b[f];
  a[idx] = v > 0.f ? v : 0.f;
}

__global__ void pool_kernel(const float* __restrict__ h, const long long* __restrict__ batch,
                            float* __restrict__ sums, float* __restrict__ cnts, int Nn) {
  int idx = blockIdx.x * blockDim.x + threadIdx.x;
  if (idx >= Nn * C) return;
  int n = idx / C, c = idx - (idx / C) * C;
  int g = (int)batch[n];
  atomicAdd(&sums[g * C + c], h[idx]);
  if (c == 0) atomicAdd(&cnts[g], 1.0f);
}

__global__ void final_kernel(const float* __restrict__ sums, const float* __restrict__ cnts,
                             const float* __restrict__ Wfc, const float* __restrict__ bfc,
                             float* __restrict__ out, int G) {
  int g = blockIdx.x * blockDim.x + threadIdx.x;
  if (g >= G) return;
  float cnt = cnts[g];
  if (cnt < 1.f) cnt = 1.f;
  float acc = 0.f;
#pragma unroll
  for (int c = 0; c < C; ++c) acc += (sums[g * C + c] / cnt) * Wfc[c];
  out[g] = acc + bfc[0];
}

extern "C" void kernel_launch(void* const* d_in, const int* in_sizes, int n_in,
                              void* d_out, int out_size, void* d_ws, size_t ws_size,
                              hipStream_t stream) {
  const float* x = (const float*)d_in[0];
  const long long* eidx = (const long long*)d_in[1];
  const long long* batch = (const long long*)d_in[2];
  const float* W1 = (const float*)d_in[3];
  const float* att_s1 = (const float*)d_in[4];
  const float* att_d1 = (const float*)d_in[5];
  const float* b1 = (const float*)d_in[6];
  const float* W2 = (const float*)d_in[7];
  const float* att_s2 = (const float*)d_in[8];
  const float* att_d2 = (const float*)d_in[9];
  const float* b2 = (const float*)d_in[10];
  const float* Wfc = (const float*)d_in[11];
  const float* bfc = (const float*)d_in[12];
  float* out = (float*)d_out;

  const int Nn = in_sizes[2];
  const int E0 = in_sizes[1] / 2;
  const int G = out_size;
  const int Etot = E0 + Nn;
  const long long* esrc = eidx;
  const long long* edst = eidx + E0;

  // workspace carve-out (256B-aligned)
  char* ws = (char*)d_ws;
  size_t off = 0;
  auto alloc = [&](size_t bytes) -> char* {
    char* p = ws + off;
    off += (bytes + 255) & ~(size_t)255;
    return p;
  };
  float* h1 = (float*)alloc((size_t)Nn * HC * 4);    // layer1 linear out; reused for layer2 h2/agg2
  float* agg1 = (float*)alloc((size_t)Nn * HC * 4);  // layer1 aggregate -> relu -> layer2 input
  float* as1 = (float*)alloc((size_t)Nn * H1 * 4);
  float* ad1 = (float*)alloc((size_t)Nn * H1 * 4);
  unsigned* m1 = (unsigned*)alloc((size_t)Nn * H1 * 4);
  float* dn1 = (float*)alloc((size_t)Nn * H1 * 4);
  float* ex1 = (float*)alloc((size_t)Etot * H1 * 4);
  __bf16* wf1 = (__bf16*)alloc((size_t)4 * 8 * 32 * 16 * 2);
  __bf16* wf2 = (__bf16*)alloc((size_t)4 * 2 * 32 * 16 * 2);
  float* sums = (float*)alloc((size_t)G * C * 4);
  float* cnts = (float*)alloc((size_t)G * 4);
  // layer2 reuses layer1 buffers (h1 region is free after agg1 is finalized)
  float* h2 = h1;
  float* agg2 = h1 + (size_t)Nn * C;
  float* as2 = as1;  float* ad2 = ad1;
  unsigned* m2 = m1; float* dn2 = dn1;
  float* ex2 = ex1;

  const int B = 256;
  const unsigned NEGINF = 0x007FFFFFu;  // f2ord(-inf) region; use f2ord(-3e38) equivalent below

  // ---- weight repack (bf16 fragments) ----
  repack_w_kernel<HC><<<CDIV(4 * 8 * 32 * 16, B), B, 0, stream>>>(W1, wf1);
  repack_w_kernel<C><<<CDIV(4 * 2 * 32 * 16, B), B, 0, stream>>>(W2, wf2);

  // ---- layer 1: h1 = x @ W1 (WMMA) ----
  wmma_gemm_kernel<HC><<<CDIV(Nn, 16), B, 0, stream>>>(x, wf1, h1, Nn);
  att_coef_kernel<<<CDIV(Nn * H1, B), B, 0, stream>>>(h1, att_s1, att_d1, as1, ad1, Nn, H1);
  fill_u32_kernel<<<CDIV(Nn * H1, B), B, 0, stream>>>(m1, NEGINF, Nn * H1);
  fill_f32_kernel<<<CDIV(Nn * H1, B), B, 0, stream>>>(dn1, 0.f, Nn * H1);
  fill_f32_kernel<<<CDIV(Nn * HC, B), B, 0, stream>>>(agg1, 0.f, Nn * HC);
  edge_max_kernel<<<CDIV(Etot * H1, B), B, 0, stream>>>(esrc, edst, E0, Nn, H1, as1, ad1, m1);
  edge_exp_kernel<<<CDIV(Etot * H1, B), B, 0, stream>>>(esrc, edst, E0, Nn, H1, as1, ad1, m1, ex1, dn1);
  edge_agg_kernel<<<CDIV(Etot * H1, B), B, 0, stream>>>(esrc, edst, E0, Nn, H1, h1, ex1, dn1, agg1);
  bias_relu_kernel<<<CDIV(Nn * HC, B), B, 0, stream>>>(agg1, b1, Nn, HC);

  // ---- layer 2: h2 = agg1 @ W2 (WMMA) ----
  wmma_gemm_kernel<C><<<CDIV(Nn, 64), B, 0, stream>>>(agg1, wf2, h2, Nn);
  att_coef_kernel<<<CDIV(Nn, B), B, 0, stream>>>(h2, att_s2, att_d2, as2, ad2, Nn, 1);
  fill_u32_kernel<<<CDIV(Nn, B), B, 0, stream>>>(m2, NEGINF, Nn);
  fill_f32_kernel<<<CDIV(Nn, B), B, 0, stream>>>(dn2, 0.f, Nn);
  fill_f32_kernel<<<CDIV(Nn * C, B), B, 0, stream>>>(agg2, 0.f, Nn * C);
  edge_max_kernel<<<CDIV(Etot, B), B, 0, stream>>>(esrc, edst, E0, Nn, 1, as2, ad2, m2);
  edge_exp_kernel<<<CDIV(Etot, B), B, 0, stream>>>(esrc, edst, E0, Nn, 1, as2, ad2, m2, ex2, dn2);
  edge_agg_kernel<<<CDIV(Etot, B), B, 0, stream>>>(esrc, edst, E0, Nn, 1, h2, ex2, dn2, agg2);
  bias_relu_kernel<<<CDIV(Nn * C, B), B, 0, stream>>>(agg2, b2, Nn, C);

  // ---- mean pool + FC ----
  fill_f32_kernel<<<CDIV(G * C, B), B, 0, stream>>>(sums, 0.f, G * C);
  fill_f32_kernel<<<CDIV(G, B), B, 0, stream>>>(cnts, 0.f, G);
  pool_kernel<<<CDIV(Nn * C, B), B, 0, stream>>>(agg2, batch, sums, cnts, Nn);
  final_kernel<<<CDIV(G, B), B, 0, stream>>>(sums, cnts, Wfc, bfc, out, G);
}
